// MoeDecoderLayerPP_75445395522230
// MI455X (gfx1250) — matrix operand order repaired and checked by reading, not compiled, not run on hardware
//
#include <hip/hip_runtime.h>

// ---------------------------------------------------------------------------
// Types / constants
// ---------------------------------------------------------------------------
typedef __attribute__((ext_vector_type(16))) __bf16 bf16x16;
typedef __attribute__((ext_vector_type(8)))  __bf16 bf16x8;
typedef __attribute__((ext_vector_type(8)))  float  f32x8;
typedef __attribute__((ext_vector_type(4)))  int    v4i;

constexpr int B_  = 2;
constexpr int S_  = 1024;
constexpr int D_  = 1024;
constexpr int H_  = 16;
constexpr int KV_ = 4;
constexpr int HD_ = 64;
constexpr int E_  = 8;
constexpr int F_  = 2048;
constexpr int T_  = B_ * S_;
constexpr float EPS_ = 1e-6f;

#define DEV __device__ __forceinline__
#define AS1 __attribute__((address_space(1)))
#define AS3 __attribute__((address_space(3)))

// CDNA5 async global->LDS copy (ASYNCcnt path), with safe fallback.
// Probe-discovered signature: (v4i AS1*, v4i AS3*, imm offset, imm cpol)
#if __has_builtin(__builtin_amdgcn_global_load_async_to_lds_b128)
#define ASYNC_CP(gsrc, ldst)                                                   \
  __builtin_amdgcn_global_load_async_to_lds_b128((AS1 v4i*)(gsrc),             \
                                                 (AS3 v4i*)(ldst), 0, 0)
#define ASYNC_WAIT() asm volatile("s_wait_asynccnt 0" ::: "memory")
#else
#define ASYNC_CP(gsrc, ldst) (*(uint4*)(ldst) = *(const uint4*)(gsrc))
#define ASYNC_WAIT() ((void)0)
#endif

DEV f32x8 zero8() {
  f32x8 z = {0.f, 0.f, 0.f, 0.f, 0.f, 0.f, 0.f, 0.f};
  return z;
}

DEV f32x8 wmma_bf16(bf16x16 a, bf16x16 b, f32x8 c) {
  // D = A(16x32 bf16) * B(32x16 bf16) + C(16x16 f32)
  return __builtin_amdgcn_wmma_f32_16x16x32_bf16(
      false, a, false, b, (short)0, c, false, false);
}

// A-fragment from a per-lane row pointer already offset by (lane>>4)*8:
// two contiguous 8-elem (16B) loads, merged.
DEV bf16x16 frag_from_rowptr(const __bf16* p) {
  bf16x8 lo = *(const bf16x8*)p;
  bf16x8 hi = *(const bf16x8*)(p + 16);
  return __builtin_shufflevector(lo, hi, 0, 1, 2, 3, 4, 5, 6, 7,
                                 8, 9, 10, 11, 12, 13, 14, 15);
}

// A-fragment: 16x32 bf16 tile (row-major, stride lda).
DEV bf16x16 frag_a(const __bf16* A, int lda) {
  int lane = threadIdx.x & 31;
  return frag_from_rowptr(A + (size_t)(lane & 15) * lda + ((lane >> 4) * 8));
}

// two 8-elem bf16 chunks at p and p+16 -> fragment (two 16B loads).
DEV bf16x16 frag_b16(const __bf16* p) {
  bf16x8 lo = *(const bf16x8*)p;
  bf16x8 hi = *(const bf16x8*)(p + 16);
  return __builtin_shufflevector(lo, hi, 0, 1, 2, 3, 4, 5, 6, 7,
                                 8, 9, 10, 11, 12, 13, 14, 15);
}

// B-fragment from N-major (transposed) storage: row = output col (lane&15),
// K-values (lane>>4)*8 + [0..7] and 16 + (lane>>4)*8 + [0..7].  ldb = K-stride of a row.
DEV bf16x16 frag_bt(const __bf16* Bt, int ldb) {
  int lane = threadIdx.x & 31;
  return frag_b16(Bt + (size_t)(lane & 15) * ldb + ((lane >> 4) * 8));
}

// ---------------------------------------------------------------------------
// fp32 -> bf16 conversion (grid-stride)
// ---------------------------------------------------------------------------
__global__ void k_f32_to_bf16(const float* __restrict__ src,
                              __bf16* __restrict__ dst, long n) {
  long i = (long)blockIdx.x * blockDim.x + threadIdx.x;
  long stride = (long)gridDim.x * blockDim.x;
  for (; i < n; i += stride) dst[i] = (__bf16)src[i];
}

// ---------------------------------------------------------------------------
// fp32 MxN -> bf16 NxM transposed convert (32x32 LDS tile), batched over z.
// ---------------------------------------------------------------------------
__global__ __launch_bounds__(256)
void k_convT(const float* __restrict__ src, __bf16* __restrict__ dst,
             int M, int N) {
  __shared__ float tile[32][33];
  const float* s = src + (size_t)blockIdx.z * M * N;
  __bf16*      d = dst + (size_t)blockIdx.z * M * N;
  int bn = blockIdx.x * 32, bm = blockIdx.y * 32;
  int tx = threadIdx.x & 31, ty = threadIdx.x >> 5;  // 32 x 8
#pragma unroll
  for (int j = 0; j < 32; j += 8)
    tile[ty + j][tx] = s[(size_t)(bm + ty + j) * N + bn + tx];
  __syncthreads();
#pragma unroll
  for (int j = 0; j < 32; j += 8)
    d[(size_t)(bn + ty + j) * M + bm + tx] = (__bf16)tile[tx][ty + j];
}

// ---------------------------------------------------------------------------
// RMSNorm: one block per token; f32 in, bf16 out
// ---------------------------------------------------------------------------
__global__ __launch_bounds__(256)
void k_rmsnorm(const float* __restrict__ x, const float* __restrict__ w,
               __bf16* __restrict__ y) {
  __shared__ float red[256];
  int t = blockIdx.x;
  const float* xr = x + (size_t)t * D_;
  float ss = 0.f;
  for (int d = threadIdx.x; d < D_; d += 256) { float v = xr[d]; ss += v * v; }
  red[threadIdx.x] = ss;
  __syncthreads();
  for (int s = 128; s > 0; s >>= 1) {
    if (threadIdx.x < s) red[threadIdx.x] += red[threadIdx.x + s];
    __syncthreads();
  }
  float rinv = rsqrtf(red[0] / (float)D_ + EPS_);
  __bf16* yr = y + (size_t)t * D_;
  for (int d = threadIdx.x; d < D_; d += 256) yr[d] = (__bf16)(xr[d] * rinv * w[d]);
}

// ---------------------------------------------------------------------------
// GEMM: C(MxN,f32) = A(MxK,bf16 row-major) @ B (passed transposed: NxK) [+res]
// 256 threads = 8 waves; 64x64 C tile; each wave: 1 A-frag x 2 B-frags.
// A/B K-tiles staged in LDS via async global->LDS b128 copies.
// ---------------------------------------------------------------------------
__global__ __launch_bounds__(256)
void k_gemm(const __bf16* __restrict__ A, const __bf16* __restrict__ Bt,
            const float* __restrict__ res, float* __restrict__ C,
            int M, int N, int Kd) {
  (void)M;
  __shared__ __attribute__((aligned(16))) __bf16 As[64 * 32];
  __shared__ __attribute__((aligned(16))) __bf16 Bs[64 * 32];
  int m0 = blockIdx.y * 64, n0 = blockIdx.x * 64;
  int tid = threadIdx.x, lane = tid & 31, wave = tid >> 5;
  int wm = (wave & 3) * 16;   // 4 row tiles of 16
  int wn = (wave >> 2) * 32;  // 2 col strips of 32
  int row = tid >> 2, seg = (tid & 3) * 8;  // 64 rows x 4 x 8-elem segments
  f32x8 acc0 = zero8(), acc1 = zero8();
  for (int kt = 0; kt < Kd; kt += 32) {
    ASYNC_CP(&A[(size_t)(m0 + row) * Kd + kt + seg], &As[row * 32 + seg]);
    ASYNC_CP(&Bt[(size_t)(n0 + row) * Kd + kt + seg], &Bs[row * 32 + seg]);
    if (kt + 32 < Kd) {
      __builtin_prefetch(&A[(size_t)(m0 + row) * Kd + kt + 32 + seg], 0, 3);
      __builtin_prefetch(&Bt[(size_t)(n0 + row) * Kd + kt + 32 + seg], 0, 3);
    }
    ASYNC_WAIT();
    __syncthreads();
    bf16x16 a  = frag_a(&As[wm * 32], 32);
    bf16x16 b0 = frag_bt(&Bs[wn * 32], 32);
    bf16x16 b1 = frag_bt(&Bs[(wn + 16) * 32], 32);
    acc0 = wmma_bf16(a, b0, acc0);
    acc1 = wmma_bf16(a, b1, acc1);
    __syncthreads();
  }
  int n = lane & 15, mb = (lane >> 4) * 8;
#pragma unroll
  for (int r = 0; r < 8; ++r) {
    size_t i0 = (size_t)(m0 + wm + mb + r) * N + n0 + wn + n;
    size_t i1 = i0 + 16;
    float v0 = acc0[r], v1 = acc1[r];
    if (res) { v0 += res[i0]; v1 += res[i1]; }
    C[i0] = v0;
    C[i1] = v1;
  }
}

// ---------------------------------------------------------------------------
// RoPE: f32 [T, nheads*HD] -> bf16 head-major [B, nheads, S, HD]
// ---------------------------------------------------------------------------
__global__ void k_rope(const float* __restrict__ qf, const int* __restrict__ pos_ids,
                       __bf16* __restrict__ qb, int nheads) {
  long total = (long)T_ * nheads * 32;
  long i = (long)blockIdx.x * blockDim.x + threadIdx.x;
  if (i >= total) return;
  int d = (int)(i & 31);
  long rem = i >> 5;
  int h = (int)(rem % nheads);
  long t = rem / nheads;
  int b = (int)(t / S_), s = (int)(t % S_);
  float p = (float)pos_ids[t];
  float ang = p * __powf(10000.f, -(float)d * (1.f / 32.f));
  float c = __cosf(ang), sn = __sinf(ang);
  const float* src = qf + (size_t)t * nheads * HD_ + h * HD_;
  float x1 = src[d], x2 = src[d + 32];
  __bf16* dst = qb + (((size_t)b * nheads + h) * S_ + s) * HD_;
  dst[d]      = (__bf16)(x1 * c - x2 * sn);
  dst[d + 32] = (__bf16)(x2 * c + x1 * sn);
}

// V: f32 [T, KV*HD] -> bf16 key-contiguous [B, KV, HD, S] (for P@V B-frags)
__global__ void k_vtrans(const float* __restrict__ vf, __bf16* __restrict__ vt) {
  long total = (long)B_ * KV_ * HD_ * S_;
  long i = (long)blockIdx.x * blockDim.x + threadIdx.x;
  if (i >= total) return;
  int s = (int)(i & (S_ - 1));
  long rem = i >> 10;
  int d = (int)(rem & (HD_ - 1));
  long rem2 = rem >> 6;
  int h = (int)(rem2 % KV_);
  int b = (int)(rem2 / KV_);
  vt[i] = (__bf16)vf[(size_t)(b * S_ + s) * (KV_ * HD_) + h * HD_ + d];
}

// ---------------------------------------------------------------------------
// Flash attention: 1 wave per 16 queries per (b,h). Causal, GQA.
// ---------------------------------------------------------------------------
__global__ __launch_bounds__(32)
void k_flash(const __bf16* __restrict__ qb, const __bf16* __restrict__ kb,
             const __bf16* __restrict__ vt, float* __restrict__ outf) {
  int bh = blockIdx.y;
  int b = bh / H_, h = bh % H_;
  int kvh = h / (H_ / KV_);
  int qi0 = blockIdx.x * 16;
  const __bf16* Q   = qb + ((size_t)bh * S_ + qi0) * HD_;
  const __bf16* Kp  = kb + ((size_t)(b * KV_ + kvh) * S_) * HD_;
  const __bf16* Vth = vt + ((size_t)(b * KV_ + kvh) * HD_) * S_;
  __shared__ __attribute__((aligned(16))) __bf16 Qs[16 * HD_];
  __shared__ __attribute__((aligned(16))) __bf16 Ps[16 * 32];
  int lane = threadIdx.x;
  for (int i = lane; i < 16 * HD_; i += 32) Qs[i] = Q[i];
  f32x8 o[4];
#pragma unroll
  for (int t = 0; t < 4; ++t) o[t] = zero8();
  float rm[8], rl[8];
#pragma unroll
  for (int r = 0; r < 8; ++r) { rm[r] = -1e30f; rl[r] = 0.f; }
  int n = lane & 15, mb = (lane >> 4) * 8;
  int kmax = qi0 + 15;
  for (int k0 = 0; k0 <= kmax; k0 += 32) {
    f32x8 s0 = zero8(), s1 = zero8();
#pragma unroll
    for (int ks = 0; ks < HD_; ks += 32) {  // scores: Q(16xHD) @ K^T
      bf16x16 a  = frag_a(&Qs[ks], HD_);
      bf16x16 b0 = frag_bt(Kp + (size_t)k0 * HD_ + ks, HD_);
      bf16x16 b1 = frag_bt(Kp + (size_t)(k0 + 16) * HD_ + ks, HD_);
      s0 = wmma_bf16(a, b0, s0);
      s1 = wmma_bf16(a, b1, s1);
    }
    float mloc[8];
#pragma unroll
    for (int r = 0; r < 8; ++r) {  // scale + causal mask
      int q = qi0 + mb + r;
      float v0 = s0[r] * 0.125f; if (k0 + n > q)      v0 = -1e30f;
      float v1 = s1[r] * 0.125f; if (k0 + 16 + n > q) v1 = -1e30f;
      s0[r] = v0; s1[r] = v1;
      mloc[r] = fmaxf(v0, v1);
    }
#pragma unroll
    for (int off = 1; off < 16; off <<= 1)
#pragma unroll
      for (int r = 0; r < 8; ++r)
        mloc[r] = fmaxf(mloc[r], __shfl_xor(mloc[r], off, 32));
    float alpha[8], psum[8];
#pragma unroll
    for (int r = 0; r < 8; ++r) {
      float mn = fmaxf(rm[r], mloc[r]);
      alpha[r] = __expf(rm[r] - mn);
      rm[r] = mn;
      float p0 = __expf(s0[r] - mn), p1 = __expf(s1[r] - mn);
      s0[r] = p0; s1[r] = p1;
      psum[r] = p0 + p1;
    }
#pragma unroll
    for (int off = 1; off < 16; off <<= 1)
#pragma unroll
      for (int r = 0; r < 8; ++r)
        psum[r] += __shfl_xor(psum[r], off, 32);
#pragma unroll
    for (int r = 0; r < 8; ++r) {
      rl[r] = rl[r] * alpha[r] + psum[r];
      Ps[(mb + r) * 32 + n]      = (__bf16)s0[r];
      Ps[(mb + r) * 32 + 16 + n] = (__bf16)s1[r];
#pragma unroll
      for (int t = 0; t < 4; ++t) o[t][r] *= alpha[r];
    }
    bf16x16 pa = frag_a(Ps, 32);  // same wave: LDS ops in-order
#pragma unroll
    for (int t = 0; t < 4; ++t) {  // O += P(16x32) @ V(32x16)
      bf16x16 vfr = frag_bt(Vth + (size_t)(t * 16) * S_ + k0, S_);
      o[t] = wmma_bf16(pa, vfr, o[t]);
    }
  }
#pragma unroll
  for (int r = 0; r < 8; ++r) {
    int q = qi0 + mb + r;
    float inv = 1.f / rl[r];
    float* op = outf + (size_t)(b * S_ + q) * (H_ * HD_) + h * HD_ + n;
#pragma unroll
    for (int t = 0; t < 4; ++t) op[t * 16] = o[t][r] * inv;
  }
}

// ---------------------------------------------------------------------------
// Router: logits = xn @ gate_w (E=8), softmax, top-2, build expert lists.
// ---------------------------------------------------------------------------
__global__ void k_zero_counts(int* counts) {
  if (threadIdx.x < E_) counts[threadIdx.x] = 0;
}

__global__ __launch_bounds__(256)
void k_router(const __bf16* __restrict__ xn, const float* __restrict__ gw,
              float* __restrict__ gate_out, int* __restrict__ counts,
              int* __restrict__ toks, float* __restrict__ wts) {
  __shared__ float red[256][E_];
  int t = blockIdx.x;
  const __bf16* x = xn + (size_t)t * D_;
  float l[E_];
#pragma unroll
  for (int e = 0; e < E_; ++e) l[e] = 0.f;
  for (int d = threadIdx.x; d < D_; d += 256) {
    float xv = (float)x[d];
#pragma unroll
    for (int e = 0; e < E_; ++e) l[e] += xv * gw[d * E_ + e];
  }
#pragma unroll
  for (int e = 0; e < E_; ++e) red[threadIdx.x][e] = l[e];
  __syncthreads();
  for (int s = 128; s > 0; s >>= 1) {
    if (threadIdx.x < s)
#pragma unroll
      for (int e = 0; e < E_; ++e) red[threadIdx.x][e] += red[threadIdx.x + s][e];
    __syncthreads();
  }
  if (threadIdx.x == 0) {
    float p[E_], mx = red[0][0];
#pragma unroll
    for (int e = 1; e < E_; ++e) mx = fmaxf(mx, red[0][e]);
    float se = 0.f;
#pragma unroll
    for (int e = 0; e < E_; ++e) { p[e] = __expf(red[0][e] - mx); se += p[e]; }
    float inv = 1.f / se;
#pragma unroll
    for (int e = 0; e < E_; ++e) { p[e] *= inv; gate_out[(size_t)t * E_ + e] = p[e]; }
    int e1 = 0;
#pragma unroll
    for (int e = 1; e < E_; ++e) if (p[e] > p[e1]) e1 = e;
    int e2 = (e1 == 0) ? 1 : 0;
#pragma unroll
    for (int e = 0; e < E_; ++e) if (e != e1 && p[e] > p[e2]) e2 = e;
    float s2 = 1.f / (p[e1] + p[e2]);
    int slot = atomicAdd(&counts[e1], 1);
    toks[e1 * T_ + slot] = t; wts[e1 * T_ + slot] = p[e1] * s2;
    slot = atomicAdd(&counts[e2], 1);
    toks[e2 * T_ + slot] = t; wts[e2 * T_ + slot] = p[e2] * s2;
  }
}

// ---------------------------------------------------------------------------
// MoE expert kernel: grid (T/16, E), 256 threads = 8 waves, gathered tokens.
// W1t/W3t: [E][F][D] (N-major), W2t: [E][D][F] (N-major).
// ---------------------------------------------------------------------------
__global__ __launch_bounds__(256)
void k_moe(const __bf16* __restrict__ xn, const __bf16* __restrict__ w1t,
           const __bf16* __restrict__ w3t, const __bf16* __restrict__ w2t,
           const int* __restrict__ counts, const int* __restrict__ toks,
           const float* __restrict__ wts, float* __restrict__ out) {
  int e = blockIdx.y;
  int cnt = counts[e];
  int tile0 = blockIdx.x * 16;
  if (tile0 >= cnt) return;
  __shared__ int   tk[16];
  __shared__ float tw[16];
  __shared__ __attribute__((aligned(16))) __bf16 Hs[16 * 1024];  // 32 KB
  int tid = threadIdx.x;
  if (tid < 16) {
    int idx = tile0 + tid;
    tk[tid] = (idx < cnt) ? toks[e * T_ + idx] : 0;
    tw[tid] = (idx < cnt) ? wts[e * T_ + idx] : 0.f;
  }
  __syncthreads();
  int lane = tid & 31, wave = tid >> 5;
  int n = lane & 15, mb = (lane >> 4) * 8;
  int half8 = (lane >> 4) * 8;
  const __bf16* W1 = w1t + (size_t)e * F_ * D_;
  const __bf16* W3 = w3t + (size_t)e * F_ * D_;
  const __bf16* W2 = w2t + (size_t)e * D_ * F_;
  const __bf16* xrow = xn + (size_t)tk[n] * D_;  // this lane's A-matrix row
  f32x8 acc[8];
#pragma unroll
  for (int i = 0; i < 8; ++i) acc[i] = zero8();
  for (int fh = 0; fh < F_; fh += 1024) {
    // phase 1: h = silu(x@W1) * (x@W3), 128 F-cols per wave
#pragma unroll 1
    for (int ft = 0; ft < 8; ++ft) {
      int fcol = fh + wave * 128 + ft * 16;
      f32x8 a1 = zero8(), a3 = zero8();
#pragma unroll 2
      for (int kd = 0; kd < D_; kd += 32) {
        bf16x16 a  = frag_from_rowptr(xrow + kd + half8);
        bf16x16 b1 = frag_bt(W1 + (size_t)fcol * D_ + kd, D_);
        bf16x16 b3 = frag_bt(W3 + (size_t)fcol * D_ + kd, D_);
        a1 = wmma_bf16(a, b1, a1);
        a3 = wmma_bf16(a, b3, a3);
      }
#pragma unroll
      for (int r = 0; r < 8; ++r) {
        float g = a1[r];
        float hv = (g / (1.f + __expf(-g))) * a3[r];  // silu(g) * a3
        Hs[(size_t)(mb + r) * 1024 + wave * 128 + ft * 16 + n] = (__bf16)hv;
      }
    }
    __syncthreads();
    // phase 2: y += h @ W2 over this F half; 128 D-cols per wave
#pragma unroll 1
    for (int dt = 0; dt < 8; ++dt) {
      int dcol = wave * 128 + dt * 16;
#pragma unroll 2
      for (int kf = 0; kf < 1024; kf += 32) {
        bf16x16 a  = frag_a(&Hs[kf], 1024);
        bf16x16 b2 = frag_bt(W2 + (size_t)dcol * F_ + fh + kf, F_);
        acc[dt] = wmma_bf16(a, b2, acc[dt]);
      }
    }
    __syncthreads();
  }
#pragma unroll 1
  for (int dt = 0; dt < 8; ++dt) {
    int dcol = wave * 128 + dt * 16;
#pragma unroll
    for (int r = 0; r < 8; ++r) {
      float w = tw[mb + r];
      if (w != 0.f)
        atomicAdd(&out[(size_t)tk[mb + r] * D_ + dcol + n], w * acc[dt][r]);
    }
  }
}

// ---------------------------------------------------------------------------
// Host orchestration
// ---------------------------------------------------------------------------
extern "C" void kernel_launch(void* const* d_in, const int* in_sizes, int n_in,
                              void* d_out, int out_size, void* d_ws, size_t ws_size,
                              hipStream_t stream) {
  (void)in_sizes; (void)n_in; (void)out_size; (void)ws_size;
  const float* hidden = (const float*)d_in[0];
  const int*   pos    = (const int*)d_in[1];
  const float* ln1w   = (const float*)d_in[3];
  const float* ln2w   = (const float*)d_in[4];
  const float* wq     = (const float*)d_in[5];
  const float* wk     = (const float*)d_in[6];
  const float* wv     = (const float*)d_in[7];
  const float* wo     = (const float*)d_in[8];
  const float* gw     = (const float*)d_in[9];
  const float* w1     = (const float*)d_in[10];
  const float* w3     = (const float*)d_in[11];
  const float* w2     = (const float*)d_in[12];

  char* wp = (char*)d_ws;
  auto alloc = [&](size_t bytes) -> void* {
    void* r = (void*)wp;
    wp += (bytes + 255) & ~(size_t)255;
    return r;
  };
  __bf16* wqt   = (__bf16*)alloc((size_t)D_ * 1024 * 2);   // [1024][D]
  __bf16* wkt   = (__bf16*)alloc((size_t)D_ * 256 * 2);    // [256][D]
  __bf16* wvt   = (__bf16*)alloc((size_t)D_ * 256 * 2);    // [256][D]
  __bf16* wot   = (__bf16*)alloc((size_t)1024 * D_ * 2);   // [D][1024]
  __bf16* w1t   = (__bf16*)alloc((size_t)E_ * D_ * F_ * 2);  // [E][F][D]
  __bf16* w3t   = (__bf16*)alloc((size_t)E_ * D_ * F_ * 2);  // [E][F][D]
  __bf16* w2t   = (__bf16*)alloc((size_t)E_ * F_ * D_ * 2);  // [E][D][F]
  __bf16* xn1   = (__bf16*)alloc((size_t)T_ * D_ * 2);
  float*  qf    = (float*) alloc((size_t)T_ * H_ * HD_ * 4);
  float*  kf    = (float*) alloc((size_t)T_ * KV_ * HD_ * 4);
  float*  vf    = (float*) alloc((size_t)T_ * KV_ * HD_ * 4);
  __bf16* qb    = (__bf16*)alloc((size_t)T_ * H_ * HD_ * 2);
  __bf16* kbuf  = (__bf16*)alloc((size_t)T_ * KV_ * HD_ * 2);
  __bf16* vtb   = (__bf16*)alloc((size_t)T_ * KV_ * HD_ * 2);
  __bf16* xn2   = (__bf16*)alloc((size_t)T_ * D_ * 2);
  int*    counts= (int*)   alloc(E_ * sizeof(int));
  int*    toks  = (int*)   alloc((size_t)E_ * T_ * sizeof(int));
  float*  wtl   = (float*) alloc((size_t)E_ * T_ * sizeof(float));
  float*  attn_f = qf;    // reuse (qf free after RoPE)
  __bf16* attn_b = xn1;   // reuse (xn1 free after QKV GEMMs)

  auto cgrid = [](long n) -> int {
    long b = (n + 255) / 256;
    return (int)(b > 16384 ? 16384 : b);
  };

  // 1. weights -> bf16, transposed to N-major for contiguous B-fragments
  k_convT<<<dim3(1024 / 32, D_ / 32, 1), 256, 0, stream>>>(wq, wqt, D_, 1024);
  k_convT<<<dim3(256 / 32, D_ / 32, 1), 256, 0, stream>>>(wk, wkt, D_, 256);
  k_convT<<<dim3(256 / 32, D_ / 32, 1), 256, 0, stream>>>(wv, wvt, D_, 256);
  k_convT<<<dim3(D_ / 32, 1024 / 32, 1), 256, 0, stream>>>(wo, wot, 1024, D_);
  k_convT<<<dim3(F_ / 32, D_ / 32, E_), 256, 0, stream>>>(w1, w1t, D_, F_);
  k_convT<<<dim3(F_ / 32, D_ / 32, E_), 256, 0, stream>>>(w3, w3t, D_, F_);
  k_convT<<<dim3(D_ / 32, F_ / 32, E_), 256, 0, stream>>>(w2, w2t, F_, D_);

  // 2. ln1
  k_rmsnorm<<<T_, 256, 0, stream>>>(hidden, ln1w, xn1);

  // 3. QKV projections (WMMA)
  k_gemm<<<dim3(1024 / 64, T_ / 64), 256, 0, stream>>>(xn1, wqt, nullptr, qf, T_, 1024, D_);
  k_gemm<<<dim3(256 / 64, T_ / 64), 256, 0, stream>>>(xn1, wkt, nullptr, kf, T_, 256, D_);
  k_gemm<<<dim3(256 / 64, T_ / 64), 256, 0, stream>>>(xn1, wvt, nullptr, vf, T_, 256, D_);

  // 4. RoPE + head-major bf16 repack
  long nq = (long)T_ * H_ * 32;
  long nk = (long)T_ * KV_ * 32;
  long nv = (long)T_ * KV_ * 64;
  k_rope<<<(int)((nq + 255) / 256), 256, 0, stream>>>(qf, pos, qb, H_);
  k_rope<<<(int)((nk + 255) / 256), 256, 0, stream>>>(kf, pos, kbuf, KV_);
  k_vtrans<<<(int)((nv + 255) / 256), 256, 0, stream>>>(vf, vtb);

  // 5. flash attention (WMMA)
  k_flash<<<dim3(S_ / 16, B_ * H_), dim3(32), 0, stream>>>(qb, kbuf, vtb, attn_f);

  // 6. O projection + residual -> d_out (h1)
  k_f32_to_bf16<<<cgrid((long)T_ * 1024), 256, 0, stream>>>(attn_f, attn_b, (long)T_ * 1024);
  float* outh = (float*)d_out;
  k_gemm<<<dim3(D_ / 64, T_ / 64), 256, 0, stream>>>(attn_b, wot, hidden, outh, T_, D_, 1024);

  // 7. ln2 + router + routed MoE (atomically accumulates into residual)
  k_rmsnorm<<<T_, 256, 0, stream>>>(outh, ln2w, xn2);
  k_zero_counts<<<1, 32, 0, stream>>>(counts);
  float* gate_out = outh + (size_t)T_ * D_;
  k_router<<<T_, 256, 0, stream>>>(xn2, gw, gate_out, counts, toks, wtl);
  k_moe<<<dim3(T_ / 16, E_), 256, 0, stream>>>(xn2, w1t, w3t, w2t, counts, toks, wtl, outh);
}
